// TinyOnnGate_54185307407176
// MI455X (gfx1250) — matrix operand order repaired and checked by reading, not compiled, not run on hardware
//
#include <hip/hip_runtime.h>
#include <hip/hip_bf16.h>
#include <stdint.h>

// Shapes from the reference: B=4, S=4096, H=4096, E=64.
#define H_DIM 4096
#define E_DIM 64
#define M_ROWS (4 * 4096)
#define KTILES (H_DIM / 32)
#define OUT_STRIDE ((size_t)M_ROWS * E_DIM)
#define EPS_NRM 1e-12f

// B-panel LDS staging: 8 k-tiles (8*32*64 bf16 = 32 KB) per chunk, double buffered.
#define CHUNK_KT 8
#define CHUNK_U4 (CHUNK_KT * 4 * 32 * 2)   // uint4 elements per chunk = 2048
#define NCHUNKS (KTILES / CHUNK_KT)        // 16

#if __has_builtin(__builtin_amdgcn_global_load_async_to_lds_b128)
#define USE_ASYNC 1
#else
#define USE_ASYNC 0
#endif

typedef __attribute__((ext_vector_type(16))) __bf16 v16bf;
typedef __attribute__((ext_vector_type(2)))  __bf16 v2bf;
typedef __attribute__((ext_vector_type(8)))  float  v8f;
typedef __attribute__((ext_vector_type(4)))  int    v4i;

#define AS1 __attribute__((address_space(1)))
#define AS3 __attribute__((address_space(3)))

union Frag {
  uint32_t u[8];
  v16bf    v;
};

// fp32 pair -> packed bf16 dword (lo = a, hi = b).
// Preferred: v_cvt_pk_bf16_f32 (RNE). Fallback: round-half-up + v_perm_b32
// (3 VALU ops; differs from RNE only on exact ties).
static __device__ __forceinline__ uint32_t pack_bf16(float a, float b) {
#if __has_builtin(__builtin_amdgcn_cvt_pk_bf16_f32)
  union { v2bf v; uint32_t u; } cv;
  cv.v = __builtin_amdgcn_cvt_pk_bf16_f32(a, b);
  return cv.u;
#else
  const uint32_t ua = __float_as_uint(a) + 0x8000u;
  const uint32_t ub = __float_as_uint(b) + 0x8000u;
  // dst bytes {b.hi16, a.hi16}: sel 2,3 from lo operand (a), 6,7 from hi (b).
  return __builtin_amdgcn_perm(ub, ua, 0x07060302u);   // v_perm_b32
#endif
}

// Cooperatively copy one 32 KB B chunk (2048 x uint4) global -> LDS.
static __device__ __forceinline__ void issue_chunk(const uint4* __restrict__ gsrc,
                                                   uint4* lbuf, int tid) {
#pragma unroll
  for (int t = 0; t < 16; ++t) {
    const int idx = tid + t * 128;
#if USE_ASYNC
    __builtin_amdgcn_global_load_async_to_lds_b128(
        (AS1 v4i*)(gsrc + idx),
        (AS3 v4i*)(lbuf + idx),
        /*offset=*/0, /*cpol=*/0);
#else
    lbuf[idx] = gsrc[idx];   // fallback: global load + ds_store
#endif
  }
}

static __device__ __forceinline__ void wait_async0() {
#if USE_ASYNC
#if __has_builtin(__builtin_amdgcn_s_wait_asynccnt)
  __builtin_amdgcn_s_wait_asynccnt(0);
#else
  asm volatile("s_wait_asynccnt 0" ::: "memory");
#endif
#endif
}

// ---------------------------------------------------------------------------
// Prep: normalize sim_matrix columns (axis 0), convert to bf16, store in the
// exact per-lane B-fragment layout for v_wmma_f32_16x16x32_bf16:
//   element (k, e): ktile kt=k/32, ntile j=e/16, n=e%16, kk=k%32
//   lane = n + 16*((kk>>3)&1) ; in-lane K index kl = (kk&7) + 8*(kk>>4) (2 per dword)
//   bf16 index = ((kt*4 + j)*32 + lane)*16 + kl
// Also: sg[e] = sigmoid(gates[e]).
// ---------------------------------------------------------------------------
__global__ void __launch_bounds__(256)
prep_wb_kernel(const float* __restrict__ sim,
               const float* __restrict__ gates,
               uint16_t* __restrict__ Wb,
               float* __restrict__ sg) {
  __shared__ float red[256];
  const int e   = blockIdx.x;       // 0..63
  const int j   = e >> 4;
  const int n   = e & 15;
  const int tid = threadIdx.x;

  float vals[16];
  float ss = 0.0f;
#pragma unroll
  for (int t = 0; t < 16; ++t) {
    const int k = tid + t * 256;
    const float v = sim[(size_t)k * E_DIM + e];
    vals[t] = v;
    ss += v * v;
  }
  red[tid] = ss;
  __syncthreads();
  for (int s = 128; s > 0; s >>= 1) {
    if (tid < s) red[tid] += red[tid + s];
    __syncthreads();
  }
  const float scale = 1.0f / fmaxf(sqrtf(red[0]), EPS_NRM);

#pragma unroll
  for (int t = 0; t < 16; ++t) {
    const int k  = tid + t * 256;
    const int kt = k >> 5;
    const int kk = k & 31;
    const int kl = (kk & 7) + ((kk >> 4) << 3);
    const int lane = n + (((kk >> 3) & 1) << 4);
    const size_t idx = (((size_t)(kt * 4 + j)) * 32 + lane) * 16 + kl;
    uint32_t u = __float_as_uint(vals[t] * scale);
    u += 0x7FFFu + ((u >> 16) & 1u);
    Wb[idx] = (uint16_t)(u >> 16);
  }
  if (tid == 0) sg[e] = 1.0f / (1.0f + __expf(-gates[e]));
}

// ---------------------------------------------------------------------------
// Main: one wave per 16-row tile (4 waves / 128-thread block). hidden_states
// is streamed once in fp32 directly to VGPRs (A-fragment lane layout),
// converted to bf16 in-register; the shared B panel is double-buffer staged
// into LDS with async global->LDS DMA overlapped with WMMA on the previous
// chunk. fp32 row sum-of-squares rides along as packed FMA chains; epilogue
// applies the row-norm scale + sigmoid/relu and writes all three outputs.
// ---------------------------------------------------------------------------
__global__ void __launch_bounds__(128)
gate_main_kernel(const float* __restrict__ hs,
                 const uint16_t* __restrict__ WbU,
                 const float* __restrict__ sg,
                 float* __restrict__ out) {
  __shared__ uint4 ldsB[2][CHUNK_U4];       // 2 x 32 KB double buffer

  const int tid  = threadIdx.x;
  const int lane = tid & 31;
  const int wave = tid >> 5;
  const int m    = lane & 15;
  const int half = lane >> 4;               // 0: K 0-7/16-23, 1: K 8-15/24-31
  const int rowBase = (blockIdx.x * 4 + wave) * 16;

  const float4* hp  = (const float4*)(hs + (size_t)(rowBase + m) * H_DIM);
  const uint4*  Wb4 = (const uint4*)WbU;

  v8f accs[4] = {v8f{}, v8f{}, v8f{}, v8f{}};
  float ss0 = 0.0f, ss1 = 0.0f, ss2 = 0.0f, ss3 = 0.0f;
  const int h2 = half * 2;                  // float4 offset inside the k-tile

  // Prologue: stage chunk 0.
  issue_chunk(Wb4, &ldsB[0][0], tid);
  wait_async0();
  __syncthreads();

  for (int c = 0; c < NCHUNKS; ++c) {
    // Overlap: kick off DMA for the next chunk into the free buffer.
    if (c + 1 < NCHUNKS)
      issue_chunk(Wb4 + (size_t)(c + 1) * CHUNK_U4, &ldsB[(c + 1) & 1][0], tid);

    const uint4* lb = &ldsB[c & 1][0];
    for (int kl = 0; kl < CHUNK_KT; ++kl) {
      const int kt = c * CHUNK_KT + kl;
      const float4 a0 = hp[kt * 8 + h2 + 0];    // K kb+8h   .. +3
      const float4 a1 = hp[kt * 8 + h2 + 1];    // K kb+8h+4 .. +7
      const float4 a2 = hp[kt * 8 + h2 + 4];    // K kb+16+8h   .. +3
      const float4 a3 = hp[kt * 8 + h2 + 5];    // K kb+16+8h+4 .. +7
      __builtin_prefetch((const void*)&hp[(kt + 4) * 8 + h2], 0, 0);

      // Row sum-of-squares: 4 independent FMA chains (pairs into v_pk_fma_f32).
      ss0 = fmaf(a0.x, a0.x, ss0); ss1 = fmaf(a0.y, a0.y, ss1);
      ss2 = fmaf(a0.z, a0.z, ss2); ss3 = fmaf(a0.w, a0.w, ss3);
      ss0 = fmaf(a1.x, a1.x, ss0); ss1 = fmaf(a1.y, a1.y, ss1);
      ss2 = fmaf(a1.z, a1.z, ss2); ss3 = fmaf(a1.w, a1.w, ss3);
      ss0 = fmaf(a2.x, a2.x, ss0); ss1 = fmaf(a2.y, a2.y, ss1);
      ss2 = fmaf(a2.z, a2.z, ss2); ss3 = fmaf(a2.w, a2.w, ss3);
      ss0 = fmaf(a3.x, a3.x, ss0); ss1 = fmaf(a3.y, a3.y, ss1);
      ss2 = fmaf(a3.z, a3.z, ss2); ss3 = fmaf(a3.w, a3.w, ss3);

      Frag A;
      A.u[0] = pack_bf16(a0.x, a0.y);
      A.u[1] = pack_bf16(a0.z, a0.w);
      A.u[2] = pack_bf16(a1.x, a1.y);
      A.u[3] = pack_bf16(a1.z, a1.w);
      A.u[4] = pack_bf16(a2.x, a2.y);
      A.u[5] = pack_bf16(a2.z, a2.w);
      A.u[6] = pack_bf16(a3.x, a3.y);
      A.u[7] = pack_bf16(a3.z, a3.w);

#pragma unroll
      for (int j = 0; j < 4; ++j) {
        const int q = ((kl * 4 + j) * 32 + lane) * 2;   // uint4 index in chunk
        const uint4 b0 = lb[q + 0];                     // ds_load_b128
        const uint4 b1 = lb[q + 1];
        Frag B;
        B.u[0] = b0.x; B.u[1] = b0.y; B.u[2] = b0.z; B.u[3] = b0.w;
        B.u[4] = b1.x; B.u[5] = b1.y; B.u[6] = b1.z; B.u[7] = b1.w;
        accs[j] = __builtin_amdgcn_wmma_f32_16x16x32_bf16(
            /*neg_a=*/false, A.v, /*neg_b=*/false, B.v,
            /*c_mod=*/(short)0, accs[j], /*reuse_a=*/false, /*reuse_b=*/false);
      }
    }

    // Drain this wave's async DMA, then fence the buffer swap block-wide.
    wait_async0();
    __syncthreads();
  }

  // Row norm: row m's elements live in lanes m and m+16 only.
  const float ss    = (ss0 + ss1) + (ss2 + ss3);
  const float tot   = ss + __shfl_xor(ss, 16);
  const float scale = 1.0f / fmaxf(sqrtf(tot), EPS_NRM);

  // C layout: VGPR i -> M = i + 8*half; N = lane%16 (+16*j).
  float rs[8];
#pragma unroll
  for (int i = 0; i < 8; ++i) rs[i] = __shfl(scale, i + half * 8);

  const int ecol = lane & 15;
#pragma unroll
  for (int j = 0; j < 4; ++j) {
    const int e = j * 16 + ecol;
    const float g = sg[e];
#pragma unroll
    for (int i = 0; i < 8; ++i) {
      const int row  = rowBase + i + half * 8;
      const size_t o = (size_t)row * E_DIM + e;
      const float raw = accs[j][i] * rs[i];
      const float sig = 1.0f / (1.0f + __expf(-raw));
      out[o]                  = raw;
      out[OUT_STRIDE + o]     = sig;
      out[2 * OUT_STRIDE + o] = fmaxf(sig - g, 0.0f);
    }
  }
}

// ---------------------------------------------------------------------------
extern "C" void kernel_launch(void* const* d_in, const int* in_sizes, int n_in,
                              void* d_out, int out_size, void* d_ws, size_t ws_size,
                              hipStream_t stream) {
  const float* hs    = (const float*)d_in[0];   // [4,4096,4096] f32
  const float* sim   = (const float*)d_in[1];   // [4096,64]     f32
  const float* gates = (const float*)d_in[2];   // [64]          f32
  float* out = (float*)d_out;                   // 3 x [16384,64] f32, concat

  // Workspace: 512 KB staged bf16 B panel + 64 gate sigmoids.
  uint16_t* Wb = (uint16_t*)d_ws;
  float*    sg = (float*)((char*)d_ws + (size_t)H_DIM * E_DIM * sizeof(uint16_t));

  prep_wb_kernel<<<dim3(E_DIM), dim3(256), 0, stream>>>(sim, gates, Wb, sg);
  gate_main_kernel<<<dim3(M_ROWS / 64), dim3(128), 0, stream>>>(hs, Wb, sg, out);
}
